// Encoder_10471130267772
// MI455X (gfx1250) — compile-verified, matrix operand order checked by reading, not run on hardware
//
#include <hip/hip_runtime.h>
#include <cstdint>

// ---------------- model constants ----------------
#define BATCH   4
#define TC      64                 // compact sequence (receptive field of 4 local-attn layers = 61)
#define MROWS   (BATCH * TC)       // 256
#define DMODEL  512
#define DMOTION 128
#define DFF     2048
#define NHEADS  8
#define DHEAD   64
#define WIN     15                 // half-window

#define FLAG_PRELU 1
#define FLAG_RES   2
#define FLAG_OBF16 4

typedef __bf16 v16bf __attribute__((ext_vector_type(16)));
typedef short  v16s  __attribute__((ext_vector_type(16)));
typedef short  v8s   __attribute__((ext_vector_type(8)));
typedef float  v8f   __attribute__((ext_vector_type(8)));
typedef unsigned short u16;

__device__ __forceinline__ u16 f2bf(float f) {
  union { float f; unsigned u; } c; c.f = f;
  unsigned u = c.u + 0x7FFFu + ((c.u >> 16) & 1u);   // round-to-nearest-even
  return (u16)(u >> 16);
}

// element e of a 16-wide bf16 fragment -> K index within the 32-K tile (ISA 7.12.2)
__device__ __forceinline__ int kmap(int e, int kh) {
  return ((e < 8) ? e : e + 8) + kh * 8;
}

// -------- one-time weight pack: fp32 K x N row-major -> bf16 B-fragment layout --------
// fragment (kt, nt): 32 lanes x 16 elems contiguous; lane: kh=lane>>4, col=nt*16+(lane&15)
__global__ void k_packW(const float* __restrict__ W, u16* __restrict__ Wp, int K, int N) {
  int g = blockIdx.x * blockDim.x + threadIdx.x;
  int total = (K >> 5) * (N >> 4) * 32;
  if (g >= total) return;
  int lane = g & 31, frag = g >> 5;
  int nt = frag % (N >> 4), kt = frag / (N >> 4);
  int kh = lane >> 4, col = nt * 16 + (lane & 15);
  u16* dst = Wp + (size_t)g * 16;
#pragma unroll
  for (int e = 0; e < 16; ++e)
    dst[e] = f2bf(W[(size_t)((kt << 5) + kmap(e, kh)) * N + col]);
}

// ---------------- bf16-WMMA GEMM, f32 accumulate ----------------
// C[M,N] = epi(A[M,K](bf16) @ Wp(packed bf16) + bias); one wave = 16(M) x 64(N) tile.
__global__ void gemm_bf16_wmma(const u16* __restrict__ A, int lda,
                               const u16* __restrict__ Wp,
                               const float* __restrict__ bias,
                               const float* __restrict__ resid,
                               const float* __restrict__ alpha,
                               float* __restrict__ C, u16* __restrict__ Cb, int ldc,
                               int M, int N, int K, int flags) {
  const int lane   = threadIdx.x & 31;
  const int wave   = threadIdx.x >> 5;
  const int wpb    = blockDim.x >> 5;
  const int tilesN = N >> 6;
  const int tilesM = (M + 15) >> 4;
  const int tileId = blockIdx.x * wpb + wave;
  if (tileId >= tilesM * tilesN) return;            // whole-wave exit, EXEC stays full
  const int tm = (tileId / tilesN) * 16;
  const int tn = (tileId % tilesN) * 64;

  const int lm = lane & 15;     // A row / B col within tile
  const int kh = lane >> 4;     // K half
  const int ntiles = N >> 4;

  v8f acc[4] = {v8f{}, v8f{}, v8f{}, v8f{}};

  int arow = tm + lm; if (arow >= M) arow = M - 1;  // clamp (stores are guarded)
  const u16* Arow = A + (size_t)arow * lda + kh * 8;

  for (int k0 = 0; k0 < K; k0 += 32) {
    // A fragment: two contiguous 16B chunks (k = kh*8+e and 16+kh*8+e)
    v8s a0 = *(const v8s*)(Arow + k0);
    v8s a1 = *(const v8s*)(Arow + k0 + 16);
    v16s as;
#pragma unroll
    for (int e = 0; e < 8; ++e) { as[e] = a0[e]; as[e + 8] = a1[e]; }
    v16bf av = __builtin_bit_cast(v16bf, as);

    const size_t ktbase = (size_t)(k0 >> 5) * ntiles + (tn >> 4);
#pragma unroll
    for (int j = 0; j < 4; ++j) {
      // packed B fragment: 16 contiguous bf16 per lane
      v16s bs = *(const v16s*)(Wp + ((ktbase + j) * 32 + lane) * 16);
      v16bf bv = __builtin_bit_cast(v16bf, bs);
      acc[j] = __builtin_amdgcn_wmma_f32_16x16x32_bf16(false, av, false, bv,
                                                       (short)0, acc[j], false, false);
    }
  }

  const float a = (flags & FLAG_PRELU) ? *alpha : 0.f;
#pragma unroll
  for (int j = 0; j < 4; ++j) {
    const int col = tn + j * 16 + lm;
    const float bcol = bias[col];
#pragma unroll
    for (int r = 0; r < 8; ++r) {
      const int row = tm + r + kh * 8;
      if (row < M) {
        float v = acc[j][r] + bcol;
        if (flags & FLAG_PRELU) v = (v >= 0.f) ? v : a * v;
        if (flags & FLAG_RES)   v += resid[(size_t)row * ldc + col];
        if (flags & FLAG_OBF16) Cb[(size_t)row * ldc + col] = f2bf(v);
        else                    C [(size_t)row * ldc + col] = v;
      }
    }
  }
}

// ---------------- compact-input assembly (x -> bf16) ----------------
__global__ void k_prep(const float* __restrict__ x, u16* __restrict__ xc) {
  int r = blockIdx.x;                  // 0..255
  int b = r >> 6, t = r & 63;
  u16* dst = xc + (size_t)r * DMOTION;
  if (t >= 2) {
    const float* src = x + ((size_t)b * 1024 + (t - 2)) * DMOTION;
    for (int c = threadIdx.x; c < DMOTION; c += blockDim.x) dst[c] = f2bf(src[c]);
  } else {
    for (int c = threadIdx.x; c < DMOTION; c += blockDim.x) dst[c] = 0;
  }
}

__global__ void k_tokens(const float* __restrict__ mu, const float* __restrict__ lv,
                         float* __restrict__ h) {
  int b = blockIdx.x;                  // 0..3
  for (int c = threadIdx.x; c < DMODEL; c += blockDim.x) {
    h[((size_t)b * TC + 0) * DMODEL + c] = mu[c];
    h[((size_t)b * TC + 1) * DMODEL + c] = lv[c];
  }
}

// ---------------- relative-position table: (31 x 64) fp32 ----------------
__global__ void k_rpe(const float* __restrict__ W1, const float* __restrict__ b1,
                      const float* __restrict__ a1,
                      const float* __restrict__ W2, const float* __restrict__ b2,
                      float* __restrict__ rel) {
  __shared__ float hb[DMODEL];
  int i = blockIdx.x;                  // 0..30
  float v = (float)(i - WIN);
  float a = *a1;
  for (int j = threadIdx.x; j < DMODEL; j += blockDim.x) {
    float t = v * W1[j] + b1[j];
    hb[j] = (t >= 0.f) ? t : a * t;
  }
  __syncthreads();
  for (int d = threadIdx.x; d < DHEAD; d += blockDim.x) {
    float s = b2[d];
    for (int j = 0; j < DMODEL; ++j) s += hb[j] * W2[(size_t)j * DHEAD + d];
    rel[i * DHEAD + d] = s;
  }
}

// ---------------- row LayerNorm over width 512 -> bf16 out (GEMM A operand) ----------------
__global__ void k_ln(const float* __restrict__ x, const float* __restrict__ g,
                     const float* __restrict__ b, u16* __restrict__ y) {
  __shared__ float red[32];
  const int r = blockIdx.x, tid = threadIdx.x;
  const float* xr = x + (size_t)r * DMODEL;
  u16*         yr = y + (size_t)r * DMODEL;
  float x0 = xr[tid], x1 = xr[tid + 256];
  float s = x0 + x1;
  for (int o = 16; o; o >>= 1) s += __shfl_xor(s, o, 32);
  if ((tid & 31) == 0) red[tid >> 5] = s;
  __syncthreads();
  if (tid == 0) { float t = 0.f; for (int i = 0; i < 8; ++i) t += red[i]; red[8] = t * (1.f / DMODEL); }
  __syncthreads();
  const float mean = red[8];
  float d0 = x0 - mean, d1 = x1 - mean;
  float vs = d0 * d0 + d1 * d1;
  for (int o = 16; o; o >>= 1) vs += __shfl_xor(vs, o, 32);
  if ((tid & 31) == 0) red[16 + (tid >> 5)] = vs;
  __syncthreads();
  if (tid == 0) { float t = 0.f; for (int i = 0; i < 8; ++i) t += red[16 + i]; red[9] = t * (1.f / DMODEL); }
  __syncthreads();
  const float rstd = rsqrtf(red[9] + 1e-5f);
  yr[tid]       = f2bf(d0 * rstd * g[tid]       + b[tid]);
  yr[tid + 256] = f2bf(d1 * rstd * g[tid + 256] + b[tid + 256]);
}

// ---------------- banded attention: one wave per (row, head), fp32 math, bf16 out ----------------
// score(i,j) = scale * q_i . (k_j + rel[j-i+15]);  softmax over |i-j|<=15
__global__ void k_attn(const float* __restrict__ q, const float* __restrict__ k,
                       const float* __restrict__ v, const float* __restrict__ rel,
                       u16* __restrict__ o) {
  const int gw   = blockIdx.x * (blockDim.x >> 5) + (threadIdx.x >> 5);
  const int lane = threadIdx.x & 31;
  const int row  = gw >> 3;           // 0..255 (exactly 2048 waves launched)
  const int head = gw & 7;
  const int b = row >> 6, t = row & 63;
  const size_t base = (size_t)row * DMODEL + head * DHEAD;
  const float q0 = q[base + lane * 2], q1 = q[base + lane * 2 + 1];
  const float scale = 0.125f;         // 1/sqrt(64)

  float sv[2 * WIN + 1];
  float m = -3.0e38f;
#pragma unroll
  for (int jj = 0; jj < 2 * WIN + 1; ++jj) {
    int j = t - WIN + jj;
    bool ok = (j >= 0) && (j < TC);
    int jc = ok ? j : 0;
    size_t kb = ((size_t)(b * TC + jc)) * DMODEL + head * DHEAD;
    float p = q0 * (k[kb + lane * 2]     + rel[jj * DHEAD + lane * 2]) +
              q1 * (k[kb + lane * 2 + 1] + rel[jj * DHEAD + lane * 2 + 1]);
    for (int off = 16; off; off >>= 1) p += __shfl_xor(p, off, 32);
    p = ok ? p * scale : -3.0e38f;
    sv[jj] = p;
    if (p > m) m = p;
  }
  float sum = 0.f;
#pragma unroll
  for (int jj = 0; jj < 2 * WIN + 1; ++jj) { float e = __expf(sv[jj] - m); sv[jj] = e; sum += e; }
  const float inv = 1.f / sum;
  float o0 = 0.f, o1 = 0.f;
#pragma unroll
  for (int jj = 0; jj < 2 * WIN + 1; ++jj) {
    int j = t - WIN + jj;
    int jc = (j >= 0 && j < TC) ? j : 0;
    size_t vb = ((size_t)(b * TC + jc)) * DMODEL + head * DHEAD;
    float w = sv[jj] * inv;
    o0 += w * v[vb + lane * 2];
    o1 += w * v[vb + lane * 2 + 1];
  }
  o[base + lane * 2]     = f2bf(o0);
  o[base + lane * 2 + 1] = f2bf(o1);
}

// gather the 8 output-feeding rows, ordered [t=0: b0..b3, t=1: b0..b3]
__global__ void k_gather(const float* __restrict__ h, float* __restrict__ g8) {
  int r = blockIdx.x;                  // 0..7
  int b = r & 3, t = r >> 2;
  const float* src = h + ((size_t)b * TC + t) * DMODEL;
  for (int c = threadIdx.x; c < DMODEL; c += blockDim.x) g8[(size_t)r * DMODEL + c] = src[c];
}

// ---------------- host side ----------------
static inline void launch_pack(const float* W, u16* Wp, int K, int N, hipStream_t s) {
  int total = (K >> 5) * (N >> 4) * 32;
  k_packW<<<(total + 255) / 256, 256, 0, s>>>(W, Wp, K, N);
}

static inline void launch_gemm(const u16* A, const u16* Wp, const float* bias,
                               const float* resid, const float* alpha,
                               float* C, u16* Cb, int M, int N, int K, int flags,
                               hipStream_t s) {
  int tiles  = ((M + 15) / 16) * (N / 64);
  int blocks = (tiles + 3) / 4;
  gemm_bf16_wmma<<<blocks, 128, 0, s>>>(A, K, Wp, bias, resid, alpha, C, Cb, N, M, N, K, flags);
}

extern "C" void kernel_launch(void* const* d_in, const int* in_sizes, int n_in,
                              void* d_out, int out_size, void* d_ws, size_t ws_size,
                              hipStream_t stream) {
  (void)in_sizes; (void)n_in; (void)out_size; (void)ws_size;
  // input order: insertion order of setup_inputs() dict, nested dicts/lists recursed
  const float* X    = (const float*)d_in[0];
  const float* MU   = (const float*)d_in[1];
  const float* LVT  = (const float*)d_in[2];
  const float* M1W  = (const float*)d_in[3];
  const float* M1B  = (const float*)d_in[4];
  const float* MA1  = (const float*)d_in[5];
  const float* M2W  = (const float*)d_in[6];
  const float* M2B  = (const float*)d_in[7];
  const float* MA2  = (const float*)d_in[8];
  const float* R1W  = (const float*)d_in[9];
  const float* R1B  = (const float*)d_in[10];
  const float* RA   = (const float*)d_in[11];
  const float* R2W  = (const float*)d_in[12];
  const float* R2B  = (const float*)d_in[13];
  const int LBASE = 14, LSTRIDE = 17;
  const float* FLG  = (const float*)d_in[LBASE + 4 * LSTRIDE + 0];
  const float* FLB  = (const float*)d_in[LBASE + 4 * LSTRIDE + 1];
  const float* D1W  = (const float*)d_in[LBASE + 4 * LSTRIDE + 2];
  const float* D1B  = (const float*)d_in[LBASE + 4 * LSTRIDE + 3];
  const float* DA   = (const float*)d_in[LBASE + 4 * LSTRIDE + 4];
  const float* D2W  = (const float*)d_in[LBASE + 4 * LSTRIDE + 5];
  const float* D2B  = (const float*)d_in[LBASE + 4 * LSTRIDE + 6];
  float* OUT = (float*)d_out;

  // ---- workspace: fp32 region then bf16 region (all sub-offsets 16B aligned) ----
  float* wsf = (float*)d_ws;
  float* h   = wsf;                       // 256*512
  float* qb  = h   + MROWS * DMODEL;
  float* kb  = qb  + MROWS * DMODEL;
  float* vb  = kb  + MROWS * DMODEL;
  float* g8  = vb  + MROWS * DMODEL;      // 8*512
  float* rel = g8  + 8 * DMODEL;          // 31*64 (pad to 2048)

  u16* wsu = (u16*)(rel + 2048);
  u16* xc  = wsu;                         // 256*128
  u16* e1  = xc  + MROWS * DMOTION;       // 256*512
  u16* tmp = e1  + MROWS * DMODEL;
  u16* ao  = tmp + MROWS * DMODEL;
  u16* ff  = ao  + MROWS * DMODEL;        // 256*2048
  u16* g8n = ff  + MROWS * DFF;           // 8*512
  u16* g8b = g8n + 8 * DMODEL;            // 8*512
  // packed weight arena
  u16* pM1 = g8b + 8 * DMODEL;            // 128*512
  u16* pM2 = pM1 + DMOTION * DMODEL;      // 512*512
  u16* pD1 = pM2 + DMODEL * DMODEL;
  u16* pD2 = pD1 + DMODEL * DMODEL;       // 512*128
  u16* pL  = pD2 + DMODEL * DMOTION;      // per layer: Q,K,V,O,W1,W2
  const size_t LQ = (size_t)DMODEL * DMODEL, LF = (size_t)DMODEL * DFF;
  const size_t LSZ = 4 * LQ + 2 * LF;

  // ---- one-time weight packing into WMMA B-fragment layout (bf16) ----
  launch_pack(M1W, pM1, DMOTION, DMODEL, stream);
  launch_pack(M2W, pM2, DMODEL, DMODEL, stream);
  launch_pack(D1W, pD1, DMODEL, DMODEL, stream);
  launch_pack(D2W, pD2, DMODEL, DMOTION, stream);
  for (int l = 0; l < 4; ++l) {
    u16* p = pL + l * LSZ;
    launch_pack((const float*)d_in[LBASE + l * LSTRIDE + 2],  p,                  DMODEL, DMODEL, stream);
    launch_pack((const float*)d_in[LBASE + l * LSTRIDE + 4],  p + LQ,             DMODEL, DMODEL, stream);
    launch_pack((const float*)d_in[LBASE + l * LSTRIDE + 6],  p + 2 * LQ,         DMODEL, DMODEL, stream);
    launch_pack((const float*)d_in[LBASE + l * LSTRIDE + 8],  p + 3 * LQ,         DMODEL, DMODEL, stream);
    launch_pack((const float*)d_in[LBASE + l * LSTRIDE + 12], p + 4 * LQ,         DMODEL, DFF,    stream);
    launch_pack((const float*)d_in[LBASE + l * LSTRIDE + 15], p + 4 * LQ + LF,    DFF,    DMODEL, stream);
  }

  // ---- motion encoder on compact 64-position window (receptive-field truncation) ----
  k_prep<<<MROWS, 128, 0, stream>>>(X, xc);
  launch_gemm(xc, pM1, M1B, nullptr, MA1, nullptr, e1, MROWS, DMODEL, DMOTION,
              FLAG_PRELU | FLAG_OBF16, stream);
  launch_gemm(e1, pM2, M2B, nullptr, MA2, h, nullptr, MROWS, DMODEL, DMODEL,
              FLAG_PRELU, stream);
  k_tokens<<<BATCH, 128, 0, stream>>>(MU, LVT, h);
  k_rpe<<<2 * WIN + 1, 64, 0, stream>>>(R1W, R1B, RA, R2W, R2B, rel);

  for (int l = 0; l < 4; ++l) {
    const float* LG = (const float*)d_in[LBASE + l * LSTRIDE + 0];
    const float* LB = (const float*)d_in[LBASE + l * LSTRIDE + 1];
    const float* QB_ = (const float*)d_in[LBASE + l * LSTRIDE + 3];
    const float* KB_ = (const float*)d_in[LBASE + l * LSTRIDE + 5];
    const float* VB_ = (const float*)d_in[LBASE + l * LSTRIDE + 7];
    const float* OB_ = (const float*)d_in[LBASE + l * LSTRIDE + 9];
    const float* FG = (const float*)d_in[LBASE + l * LSTRIDE + 10];
    const float* FB = (const float*)d_in[LBASE + l * LSTRIDE + 11];
    const float* B1 = (const float*)d_in[LBASE + l * LSTRIDE + 13];
    const float* FA = (const float*)d_in[LBASE + l * LSTRIDE + 14];
    const float* B2 = (const float*)d_in[LBASE + l * LSTRIDE + 16];
    u16* p = pL + l * LSZ;

    // attention block (pre-LN)
    k_ln<<<MROWS, 256, 0, stream>>>(h, LG, LB, tmp);
    launch_gemm(tmp, p,          QB_, nullptr, nullptr, qb, nullptr, MROWS, DMODEL, DMODEL, 0, stream);
    launch_gemm(tmp, p + LQ,     KB_, nullptr, nullptr, kb, nullptr, MROWS, DMODEL, DMODEL, 0, stream);
    launch_gemm(tmp, p + 2 * LQ, VB_, nullptr, nullptr, vb, nullptr, MROWS, DMODEL, DMODEL, 0, stream);
    k_attn<<<(MROWS * NHEADS) / 4, 128, 0, stream>>>(qb, kb, vb, rel, ao);
    launch_gemm(ao, p + 3 * LQ,  OB_, h, nullptr, h, nullptr, MROWS, DMODEL, DMODEL, FLAG_RES, stream);

    // FFN block (pre-LN)
    k_ln<<<MROWS, 256, 0, stream>>>(h, FG, FB, tmp);
    launch_gemm(tmp, p + 4 * LQ, B1, nullptr, FA, nullptr, ff, MROWS, DFF, DMODEL,
                FLAG_PRELU | FLAG_OBF16, stream);
    launch_gemm(ff, p + 4 * LQ + LF, B2, h, nullptr, h, nullptr, MROWS, DMODEL, DFF, FLAG_RES, stream);
  }

  // ---- decoder runs on just the 8 rows that reach the output ----
  k_gather<<<8, 128, 0, stream>>>(h, g8);
  k_ln<<<8, 256, 0, stream>>>(g8, FLG, FLB, g8n);
  launch_gemm(g8n, pD1, D1B, nullptr, DA, nullptr, g8b, 8, DMODEL, DMODEL,
              FLAG_PRELU | FLAG_OBF16, stream);
  launch_gemm(g8b, pD2, D2B, nullptr, nullptr, OUT, nullptr, 8, DMOTION, DMODEL, 0, stream);
}
